// DKVMN_39032662786544
// MI455X (gfx1250) — matrix operand order repaired and checked by reading, not compiled, not run on hardware
//
#include <hip/hip_runtime.h>
#include <hip/hip_bf16.h>
#include <math.h>

#define NUM_C  200
#define EMB    128
#define SIZE_M 50
#define N_STU  4000
#define BATCH  64
#define SEQL   200
#define NTOK   (BATCH * SEQL)

typedef __attribute__((ext_vector_type(16))) __bf16 v16bf;
typedef __attribute__((ext_vector_type(8)))  __bf16 v8bf;
typedef __attribute__((ext_vector_type(8)))  float  v8f;

static __device__ __forceinline__ __bf16 f2bf(float f) {
    __hip_bfloat16 h = __float2bfloat16(f);
    return *reinterpret_cast<__bf16*>(&h);
}
static __device__ __forceinline__ float sigm(float x) { return 1.0f / (1.0f + __expf(-x)); }

// ---------------------------------------------------------------- converts
__global__ void cvt_bf16(const float* __restrict__ in, __bf16* __restrict__ out, int n) {
    int i = blockIdx.x * blockDim.x + threadIdx.x;
    int stride = gridDim.x * blockDim.x;
    for (; i < n; i += stride) out[i] = f2bf(in[i]);
}

// in: K x N row-major (fp32) -> out: N x K row-major (bf16)
__global__ void cvt_bf16_T(const float* __restrict__ in, __bf16* __restrict__ out, int K, int N) {
    int i = blockIdx.x * blockDim.x + threadIdx.x;
    int total = K * N;
    int stride = gridDim.x * blockDim.x;
    for (; i < total; i += stride) {
        int n = i / K, k = i % K;
        out[i] = f2bf(in[(size_t)k * N + n]);
    }
}

// ---------------------------------------------------------------- WMMA GEMM
// C(MxN,f32) = act( A(MxK,bf16) @ BT(NxK,bf16)^T + bias [+ C if ACCUM] )
// ACT: 0=none 1=relu 2=sigmoid 3=tanh. K must be a multiple of 32.
// Block = 128 threads (4 waves); wave -> 16x64 output strip; block -> 64x64.
template <int ACT, int ACCUM>
__global__ void gemm_wmma(const __bf16* __restrict__ A, const __bf16* __restrict__ BT,
                          const float* __restrict__ bias, float* __restrict__ C,
                          int M, int N, int K) {
    const int lane = threadIdx.x & 31;
    const int wave = threadIdx.x >> 5;
    const int m0 = blockIdx.y * 64 + wave * 16;
    const int n0 = blockIdx.x * 64;

    const int arow = min(m0 + (lane & 15), M - 1);
    const int kba  = (lane >> 4) * 16;   // A: 16 contiguous k values (two 8-elem chunks)
    const int kbb  = (lane >> 4) * 16;   // B: 16 contiguous k values
    const int cb   = n0 + (lane & 15);

    const __bf16* Arow = A + (size_t)arow * K;
    const __bf16* B0 = BT + (size_t)min(cb +  0, N - 1) * K;
    const __bf16* B1 = BT + (size_t)min(cb + 16, N - 1) * K;
    const __bf16* B2 = BT + (size_t)min(cb + 32, N - 1) * K;
    const __bf16* B3 = BT + (size_t)min(cb + 48, N - 1) * K;

    v8f acc0 = {}, acc1 = {}, acc2 = {}, acc3 = {};

    for (int k0 = 0; k0 < K; k0 += 32) {
        v8bf a_lo = *reinterpret_cast<const v8bf*>(Arow + k0 + kba);
        v8bf a_hi = *reinterpret_cast<const v8bf*>(Arow + k0 + kba + 8);
        v16bf av;
#pragma unroll
        for (int i = 0; i < 8; ++i) { av[i] = a_lo[i]; av[i + 8] = a_hi[i]; }
        v16bf b0 = *reinterpret_cast<const v16bf*>(B0 + k0 + kbb);
        v16bf b1 = *reinterpret_cast<const v16bf*>(B1 + k0 + kbb);
        v16bf b2 = *reinterpret_cast<const v16bf*>(B2 + k0 + kbb);
        v16bf b3 = *reinterpret_cast<const v16bf*>(B3 + k0 + kbb);
        acc0 = __builtin_amdgcn_wmma_f32_16x16x32_bf16(false, av, false, b0, (short)0, acc0, false, false);
        acc1 = __builtin_amdgcn_wmma_f32_16x16x32_bf16(false, av, false, b1, (short)0, acc1, false, false);
        acc2 = __builtin_amdgcn_wmma_f32_16x16x32_bf16(false, av, false, b2, (short)0, acc2, false, false);
        acc3 = __builtin_amdgcn_wmma_f32_16x16x32_bf16(false, av, false, b3, (short)0, acc3, false, false);
    }

    auto store_tile = [&](v8f acc, int tn0) {
        int col = tn0 + (lane & 15);
        if (col >= N) return;
        float bv = bias ? bias[col] : 0.0f;
#pragma unroll
        for (int j = 0; j < 8; ++j) {
            int row = m0 + ((lane >> 4) << 3) + j;
            if (row >= M) continue;
            size_t idx = (size_t)row * N + col;
            float v = acc[j] + bv;
            if (ACCUM) v += C[idx];
            if (ACT == 1) v = fmaxf(v, 0.0f);
            else if (ACT == 2) v = sigm(v);
            else if (ACT == 3) v = tanhf(v);
            C[idx] = v;
        }
    };
    store_tile(acc0, n0);
    store_tile(acc1, n0 + 16);
    store_tile(acc2, n0 + 32);
    store_tile(acc3, n0 + 48);
}

// ---------------------------------------------------------------- gather + gates
// one block (128 threads) per token; computes gated k and v
__global__ void gather_gate(const int* __restrict__ student, const int* __restrict__ skill,
                            const int* __restrict__ answer, const float* __restrict__ STU,
                            const float* __restrict__ k_emb, const float* __restrict__ v_emb,
                            const float* __restrict__ Wgk, const float* __restrict__ bgk,
                            const float* __restrict__ Wgv, const float* __restrict__ bgv,
                            float* __restrict__ kout, float* __restrict__ vout) {
    __shared__ float rk[EMB], rv[EMB];
    int t = blockIdx.x;
    int d = threadIdx.x;
    int stu = student[t] - 1;
    int sk = skill[t];
    int ans = answer[t];
    int ax = (ans == 2) ? 1 : ans;
    int x = sk + NUM_C * ax;
    float sh = STU[(size_t)stu * EMB + d];
    float kk = k_emb[(size_t)sk * EMB + d];
    float vv = v_emb[(size_t)x * EMB + d];
    rk[d] = sh * Wgk[d] + kk * Wgk[EMB + d];
    rv[d] = sh * Wgv[d] + vv * Wgv[EMB + d];
    __syncthreads();
    for (int s = 64; s > 0; s >>= 1) {
        if (d < s) { rk[d] += rk[d + s]; rv[d] += rv[d + s]; }
        __syncthreads();
    }
    float gk = sigm(rk[0] + bgk[0]);
    float gv = sigm(rv[0] + bgv[0]);
    kout[(size_t)t * EMB + d] = gk * sh + (1.0f - gk) * kk;
    vout[(size_t)t * EMB + d] = gv * sh + (1.0f - gv) * vv;
}

// ---------------------------------------------------------------- softmax over SIZE_M
__global__ void softmax50(float* __restrict__ w) {
    __shared__ float s[64];
    int t = blockIdx.x, i = threadIdx.x;
    float v = (i < SIZE_M) ? w[(size_t)t * SIZE_M + i] : -1e30f;
    s[i] = v;
    __syncthreads();
    for (int st = 32; st > 0; st >>= 1) { if (i < st) s[i] = fmaxf(s[i], s[i + st]); __syncthreads(); }
    float mx = s[0];
    __syncthreads();
    float ex = (i < SIZE_M) ? __expf(v - mx) : 0.0f;
    s[i] = ex;
    __syncthreads();
    for (int st = 32; st > 0; st >>= 1) { if (i < st) s[i] += s[i + st]; __syncthreads(); }
    if (i < SIZE_M) w[(size_t)t * SIZE_M + i] = ex / s[0];
}

// ---------------------------------------------------------------- sequential memory scan
// grid = BATCH, block = EMB; thread d owns Mv[:,d] in registers
__global__ void dkvmn_scan(const float* __restrict__ w, const float* __restrict__ e,
                           const float* __restrict__ a, const float* __restrict__ Mv0,
                           float* __restrict__ read) {
    __shared__ float ws_[SIZE_M];
    int b = blockIdx.x, d = threadIdx.x;
    float Mv[SIZE_M];
#pragma unroll
    for (int m = 0; m < SIZE_M; ++m) Mv[m] = Mv0[m * EMB + d];
    for (int t = 0; t < SEQL; ++t) {
        size_t tok = (size_t)b * SEQL + t;
        if (d < SIZE_M) ws_[d] = w[tok * SIZE_M + d];
        __syncthreads();
        float ed = e[tok * EMB + d];
        float ad = a[tok * EMB + d];
        float r = 0.0f;
#pragma unroll
        for (int m = 0; m < SIZE_M; ++m) {
            float wm = ws_[m];
            r += wm * Mv[m];                 // read uses pre-update Mv
            Mv[m] += wm * (ad - ed * Mv[m]); // Mv*(1-w*e) + w*a
        }
        read[tok * EMB + d] = r;
        __syncthreads();
    }
}

// ---------------------------------------------------------------- final gathered prediction
// one wave per (b, l<L-1): pred = sigmoid(f[b,l,:] . Wp[:,nc] + bp[nc])
__global__ void pred_kernel(const float* __restrict__ f, const int* __restrict__ skill,
                            const float* __restrict__ Wp, const float* __restrict__ bp,
                            float* __restrict__ out) {
    int idx = blockIdx.x;
    int b = idx / (SEQL - 1), l = idx % (SEQL - 1);
    int ns = skill[b * SEQL + l + 1];
    int nc = min(max(ns, 0), NUM_C - 1);
    int lane = threadIdx.x;
    const float* fv = f + ((size_t)b * SEQL + l) * EMB;
    float p = 0.0f;
    for (int d = lane; d < EMB; d += 32) p += fv[d] * Wp[(size_t)d * NUM_C + nc];
#pragma unroll
    for (int s = 16; s > 0; s >>= 1) p += __shfl_down(p, s, 32);
    if (lane == 0) {
        float pr = sigm(p + bp[nc]);
        out[idx] = (ns == NUM_C) ? 0.0f : pr;
    }
}

// ---------------------------------------------------------------- launcher
extern "C" void kernel_launch(void* const* d_in, const int* in_sizes, int n_in,
                              void* d_out, int out_size, void* d_ws, size_t ws_size,
                              hipStream_t stream) {
    (void)in_sizes; (void)n_in; (void)out_size; (void)ws_size;
    const int*   student   = (const int*)d_in[0];
    const int*   skill     = (const int*)d_in[1];
    const int*   answer    = (const int*)d_in[2];
    const float* G         = (const float*)d_in[3];
    const float* stu_table = (const float*)d_in[4];
    const float* W1        = (const float*)d_in[5];
    const float* b1        = (const float*)d_in[6];
    const float* W2        = (const float*)d_in[7];
    const float* b2        = (const float*)d_in[8];
    const float* k_emb     = (const float*)d_in[9];
    const float* v_emb     = (const float*)d_in[10];
    const float* Mk        = (const float*)d_in[11];
    const float* Mv0       = (const float*)d_in[12];
    const float* Wgk       = (const float*)d_in[13];
    const float* bgk       = (const float*)d_in[14];
    const float* Wgv       = (const float*)d_in[15];
    const float* bgv       = (const float*)d_in[16];
    const float* We        = (const float*)d_in[17];
    const float* be        = (const float*)d_in[18];
    const float* Wa        = (const float*)d_in[19];
    const float* ba        = (const float*)d_in[20];
    const float* Wf        = (const float*)d_in[21];
    const float* bfb       = (const float*)d_in[22];
    const float* Wp        = (const float*)d_in[23];
    const float* bp        = (const float*)d_in[24];

    char* wsb = (char*)d_ws;
    size_t off = 0;
    auto alloc = [&](size_t bytes) -> char* {
        char* p = wsb + off;
        off = (off + bytes + 255) & ~(size_t)255;
        return p;
    };

    __bf16* Gbf    = (__bf16*)alloc((size_t)N_STU * N_STU * 2);
    __bf16* Sbf    = (__bf16*)alloc((size_t)N_STU * EMB * 2);
    __bf16* W1T    = (__bf16*)alloc(EMB * EMB * 2);
    __bf16* W2T    = (__bf16*)alloc(EMB * EMB * 2);
    __bf16* WeT    = (__bf16*)alloc(EMB * EMB * 2);
    __bf16* WaT    = (__bf16*)alloc(EMB * EMB * 2);
    __bf16* WfTopT = (__bf16*)alloc(EMB * EMB * 2);
    __bf16* WfBotT = (__bf16*)alloc(EMB * EMB * 2);
    __bf16* Mkbf   = (__bf16*)alloc(SIZE_M * EMB * 2);
    float*  T1     = (float*)alloc((size_t)N_STU * EMB * 4);
    __bf16* T1bT   = (__bf16*)alloc((size_t)N_STU * EMB * 2);
    float*  H      = (float*)alloc((size_t)N_STU * EMB * 4);
    __bf16* Hbf    = (__bf16*)alloc((size_t)N_STU * EMB * 2);
    float*  T2     = (float*)alloc((size_t)N_STU * EMB * 4);
    __bf16* T2bT   = (__bf16*)alloc((size_t)N_STU * EMB * 2);
    float*  STU    = (float*)alloc((size_t)N_STU * EMB * 4);
    float*  kbuf   = (float*)alloc((size_t)NTOK * EMB * 4);
    float*  vbuf   = (float*)alloc((size_t)NTOK * EMB * 4);
    __bf16* kbf    = (__bf16*)alloc((size_t)NTOK * EMB * 2);
    __bf16* vbf    = (__bf16*)alloc((size_t)NTOK * EMB * 2);
    float*  wbuf   = (float*)alloc((size_t)NTOK * SIZE_M * 4);
    float*  ebuf   = (float*)alloc((size_t)NTOK * EMB * 4);
    float*  abuf   = (float*)alloc((size_t)NTOK * EMB * 4);
    float*  rdbuf  = (float*)alloc((size_t)NTOK * EMB * 4);
    __bf16* rdbf   = (__bf16*)alloc((size_t)NTOK * EMB * 2);
    float*  fbuf   = (float*)alloc((size_t)NTOK * EMB * 4);

    auto cblocks = [](size_t n) { return (unsigned)((n + 255) / 256); };

    // bf16 conversions of static operands
    cvt_bf16<<<8192, 256, 0, stream>>>(G, Gbf, N_STU * N_STU);
    cvt_bf16<<<cblocks((size_t)N_STU * EMB), 256, 0, stream>>>(stu_table, Sbf, N_STU * EMB);
    cvt_bf16_T<<<64, 256, 0, stream>>>(W1, W1T, EMB, EMB);
    cvt_bf16_T<<<64, 256, 0, stream>>>(W2, W2T, EMB, EMB);
    cvt_bf16_T<<<64, 256, 0, stream>>>(We, WeT, EMB, EMB);
    cvt_bf16_T<<<64, 256, 0, stream>>>(Wa, WaT, EMB, EMB);
    cvt_bf16_T<<<64, 256, 0, stream>>>(Wf, WfTopT, EMB, EMB);                 // rows 0..127
    cvt_bf16_T<<<64, 256, 0, stream>>>(Wf + EMB * EMB, WfBotT, EMB, EMB);     // rows 128..255
    cvt_bf16<<<cblocks(SIZE_M * EMB), 256, 0, stream>>>(Mk, Mkbf, SIZE_M * EMB); // Mk already NxK

    dim3 blk(128);
    dim3 gS((EMB + 63) / 64, (N_STU + 63) / 64);   // 4000 x 128 outputs
    dim3 gT((EMB + 63) / 64, (NTOK + 63) / 64);    // 12800 x 128 outputs
    dim3 gW((SIZE_M + 63) / 64, (NTOK + 63) / 64); // 12800 x 50 outputs

    // stu_emb pipeline: T1 = stu_table@W1+b1 ; H = relu(G@T1) ; T2 = H@W2+b2 ; STU = G@T2
    gemm_wmma<0, 0><<<gS, blk, 0, stream>>>(Sbf, W1T, b1, T1, N_STU, EMB, EMB);
    cvt_bf16_T<<<cblocks((size_t)N_STU * EMB), 256, 0, stream>>>(T1, T1bT, N_STU, EMB);
    gemm_wmma<1, 0><<<gS, blk, 0, stream>>>(Gbf, T1bT, nullptr, H, N_STU, EMB, N_STU);
    cvt_bf16<<<cblocks((size_t)N_STU * EMB), 256, 0, stream>>>(H, Hbf, N_STU * EMB);
    gemm_wmma<0, 0><<<gS, blk, 0, stream>>>(Hbf, W2T, b2, T2, N_STU, EMB, EMB);
    cvt_bf16_T<<<cblocks((size_t)N_STU * EMB), 256, 0, stream>>>(T2, T2bT, N_STU, EMB);
    gemm_wmma<0, 0><<<gS, blk, 0, stream>>>(Gbf, T2bT, nullptr, STU, N_STU, EMB, N_STU);

    // per-token gating
    gather_gate<<<NTOK, EMB, 0, stream>>>(student, skill, answer, STU, k_emb, v_emb,
                                          Wgk, bgk, Wgv, bgv, kbuf, vbuf);
    cvt_bf16<<<cblocks((size_t)NTOK * EMB), 256, 0, stream>>>(kbuf, kbf, NTOK * EMB);
    cvt_bf16<<<cblocks((size_t)NTOK * EMB), 256, 0, stream>>>(vbuf, vbf, NTOK * EMB);

    // attention weights, erase, add
    gemm_wmma<0, 0><<<gW, blk, 0, stream>>>(kbf, Mkbf, nullptr, wbuf, NTOK, SIZE_M, EMB);
    softmax50<<<NTOK, 64, 0, stream>>>(wbuf);
    gemm_wmma<2, 0><<<gT, blk, 0, stream>>>(vbf, WeT, be, ebuf, NTOK, EMB, EMB);
    gemm_wmma<3, 0><<<gT, blk, 0, stream>>>(vbf, WaT, ba, abuf, NTOK, EMB, EMB);

    // sequential memory update + read
    dkvmn_scan<<<BATCH, EMB, 0, stream>>>(wbuf, ebuf, abuf, Mv0, rdbuf);

    // f = tanh([read,k] @ Wf + bf), split as read@Wf_top then (+ k@Wf_bot, tanh)
    cvt_bf16<<<cblocks((size_t)NTOK * EMB), 256, 0, stream>>>(rdbuf, rdbf, NTOK * EMB);
    gemm_wmma<0, 0><<<gT, blk, 0, stream>>>(rdbf, WfTopT, bfb, fbuf, NTOK, EMB, EMB);
    gemm_wmma<3, 1><<<gT, blk, 0, stream>>>(kbf, WfBotT, nullptr, fbuf, NTOK, EMB, EMB);

    // gathered prediction (only the needed column of p)
    pred_kernel<<<BATCH * (SEQL - 1), 32, 0, stream>>>(fbuf, skill, Wp, bp, (float*)d_out);
}
